// ParallelTransformer_21354577396202
// MI455X (gfx1250) — compile-verified, matrix operand order checked by reading, not run on hardware
//
#include <hip/hip_runtime.h>
#include <math.h>

// ---------------------------------------------------------------------------
// Types
// ---------------------------------------------------------------------------
typedef __bf16 bf16;
typedef __attribute__((ext_vector_type(16))) __bf16 v16bf;
typedef __attribute__((ext_vector_type(8)))  float  v8f;
typedef __attribute__((ext_vector_type(4)))  int    v4i;

// Model constants
#define NB    8      // batch
#define SQ    512    // sequence
#define DIM   768
#define NH    12
#define DH    64
#define HID   3072
#define BS    4096   // NB*SQ
#define BH    96     // NB*NH

enum { OUT_F32 = 0, OUT_BF16 = 1, OUT_QK = 2, OUT_VT = 3, OUT_CTX = 4 };

// ---------------------------------------------------------------------------
// CDNA5 async global->LDS staging (ASYNCcnt path), with builtin/asm fallbacks.
// Builtin signature (from hipcc diagnostic): param0 = addrspace(1) v4i*.
// ---------------------------------------------------------------------------
typedef __attribute__((address_space(1))) v4i gv4i;
typedef __attribute__((address_space(3))) v4i lv4i;

__device__ __forceinline__ void async_ld_b128(const void* g, void* l)
{
#if __has_builtin(__builtin_amdgcn_global_load_async_to_lds_b128)
    __builtin_amdgcn_global_load_async_to_lds_b128(
        (gv4i*)g, (lv4i*)l, 0, 0);
#else
    unsigned lo = (unsigned)(unsigned long long)l;   // flat LDS addr[31:0] = LDS offset
    asm volatile("global_load_async_to_lds_b128 %0, %1, off"
                 :: "v"(lo), "v"(g) : "memory");
#endif
}

template <int N>
__device__ __forceinline__ void wait_asynccnt()
{
#if __has_builtin(__builtin_amdgcn_s_wait_asynccnt)
    __builtin_amdgcn_s_wait_asynccnt((unsigned short)N);
#else
    asm volatile("s_wait_asynccnt %0" :: "n"(N) : "memory");
#endif
}

// ---------------------------------------------------------------------------
// WMMA NT GEMM:  C[z][m][n] = act( (sum_k A[m][k]*Bt[n][k] + bias[n]) * alpha )
// A: bf16 [M x K] row-major (lda), Bt: bf16 [N x K] row-major (ldb).
// Block tile 128x64, 8 waves (4x2), each wave 32x32 = 2x2 WMMA accumulators.
// Double-buffered LDS fed by GLOBAL_LOAD_ASYNC_TO_LDS_B128 (3 ops/thread/tile,
// in-order ASYNCcnt => s_wait_asynccnt 3 retires the previous tile).
// ---------------------------------------------------------------------------
__global__ __launch_bounds__(256) void gemm_nt(
    const bf16* __restrict__ A, const bf16* __restrict__ Bt,
    const float* __restrict__ bias,
    float* __restrict__ outF, bf16* __restrict__ outB,
    int M, int N, int K, int lda, int ldb, int ldo,
    long sA, long sB, long sC,
    float alpha, int act, int outmode)
{
    const int z = blockIdx.z;
    A  += (long)z * sA;
    Bt += (long)z * sB;

    __shared__ __align__(16) bf16 As[2][128 * 40];  // 32 cols + 8 pad (16B-aligned rows)
    __shared__ __align__(16) bf16 Bs[2][64 * 40];

    const int tid  = threadIdx.x;
    const int lane = tid & 31;
    const int wid  = tid >> 5;
    const int wm   = wid & 3;        // wave M index (0..3)
    const int wn   = wid >> 2;       // wave N index (0..1)
    const int m0   = blockIdx.y * 128;
    const int n0   = blockIdx.x * 64;

    const int r  = lane & 15;        // row-in-tile for A/B operands
    const int hl = lane >> 4;        // half-wave select
    const int c0 = hl * 8;           // K chunk start: lanes<16 -> {0..7,16..23}

    // Per-thread staging coordinates
    const int arow0 = (tid) >> 2,        acol0 = ((tid) & 3) * 8;        // A chunk 0
    const int arow1 = (tid + 256) >> 2,  acol1 = ((tid + 256) & 3) * 8;  // A chunk 1
    const int brow  = tid >> 2,          bcol  = (tid & 3) * 8;          // B chunk
    int agr0 = m0 + arow0; if (agr0 >= M) agr0 = M - 1;                  // clamp small-M
    int agr1 = m0 + arow1; if (agr1 >= M) agr1 = M - 1;

    v8f acc[2][2] = {};

    // Prologue: stream tile 0 into buffer 0
    async_ld_b128(A  + (long)agr0 * lda + acol0,        &As[0][arow0 * 40 + acol0]);
    async_ld_b128(A  + (long)agr1 * lda + acol1,        &As[0][arow1 * 40 + acol1]);
    async_ld_b128(Bt + (long)(n0 + brow) * ldb + bcol,  &Bs[0][brow * 40 + bcol]);

    int buf = 0;
    for (int kt = 0; kt < K; kt += 32) {
        __syncthreads();   // all waves done computing on buf^1: safe to overwrite
        const bool has_next = (kt + 32) < K;
        if (has_next) {
            const int kn = kt + 32, nb = buf ^ 1;
            async_ld_b128(A  + (long)agr0 * lda + kn + acol0,       &As[nb][arow0 * 40 + acol0]);
            async_ld_b128(A  + (long)agr1 * lda + kn + acol1,       &As[nb][arow1 * 40 + acol1]);
            async_ld_b128(Bt + (long)(n0 + brow) * ldb + kn + bcol, &Bs[nb][brow * 40 + bcol]);
            if (kt + 64 < K) {  // keepahead prefetch (global_prefetch_b8)
                __builtin_prefetch(A  + (long)agr0 * lda + kt + 64, 0, 1);
                __builtin_prefetch(Bt + (long)(n0 + brow) * ldb + kt + 64, 0, 1);
            }
            wait_asynccnt<3>();   // 3 oldest (current tile) retired; next tile in flight
        } else {
            wait_asynccnt<0>();
        }
        __syncthreads();   // current tile visible to all waves

        // Fragments per ISA 16-bit A/B layout:
        // lanes 0-15: row=lane,    K chunks at {0,16}
        // lanes16-31: row=lane-16, K chunks at {8,24}
        union Fr { v16bf v; uint4 q[2]; };
        Fr fa[2], fb[2];
        #pragma unroll
        for (int mi = 0; mi < 2; ++mi) {
            int ar = wm * 32 + mi * 16 + r;
            fa[mi].q[0] = *(const uint4*)(&As[buf][ar * 40 + c0]);
            fa[mi].q[1] = *(const uint4*)(&As[buf][ar * 40 + c0 + 16]);
        }
        #pragma unroll
        for (int ni = 0; ni < 2; ++ni) {
            int br = wn * 32 + ni * 16 + r;
            fb[ni].q[0] = *(const uint4*)(&Bs[buf][br * 40 + c0]);
            fb[ni].q[1] = *(const uint4*)(&Bs[buf][br * 40 + c0 + 16]);
        }
        #pragma unroll
        for (int mi = 0; mi < 2; ++mi)
            #pragma unroll
            for (int ni = 0; ni < 2; ++ni)
                acc[mi][ni] = __builtin_amdgcn_wmma_f32_16x16x32_bf16(
                    false, fa[mi].v, false, fb[ni].v,
                    (short)0, acc[mi][ni], false, false);
        buf ^= 1;
    }

    // Epilogue. C layout: VGPR e holds M = e (lanes<16) / e+8 (lanes>=16), N = lane&15.
    #pragma unroll
    for (int mi = 0; mi < 2; ++mi)
      #pragma unroll
      for (int ni = 0; ni < 2; ++ni)
        #pragma unroll
        for (int e = 0; e < 8; ++e) {
            int row = m0 + wm * 32 + mi * 16 + e + hl * 8;
            int col = n0 + wn * 32 + ni * 16 + (lane & 15);
            if (row >= M) continue;
            float v = acc[mi][ni][e];
            if (bias) v += bias[col];
            v *= alpha;
            if (act) v = 0.5f * v * (1.0f + erff(v * 0.70710678118654752f)); // exact GELU
            switch (outmode) {
            case OUT_F32:
                outF[(long)z * sC + (long)row * ldo + col] = v; break;
            case OUT_BF16:
                outB[(long)z * sC + (long)row * ldo + col] = (bf16)v; break;
            case OUT_QK: {  // [BS x 768] -> [B,H,S,DH]
                int b = row >> 9, s = row & 511, h = col >> 6, d = col & 63;
                outB[(((long)(b * NH + h)) * SQ + s) * DH + d] = (bf16)v;
            } break;
            case OUT_VT: {  // [BS x 768] -> [B,H,DH,S]
                int b = row >> 9, s = row & 511, h = col >> 6, d = col & 63;
                outB[(((long)(b * NH + h)) * DH + d) * SQ + s] = (bf16)v;
            } break;
            case OUT_CTX: { // z=(b,h): ctx[b, row, h*64+col] in [BS x 768]
                int b = z / NH, h = z % NH;
                outB[((long)(b * SQ + row)) * DIM + h * DH + col] = (bf16)v;
            } break;
            }
        }
}

// ---------------------------------------------------------------------------
// Weight transpose + fp32->bf16:  Wt[n][k] = (bf16)W[k][n]
// ---------------------------------------------------------------------------
__global__ __launch_bounds__(256) void transp_k(
    const float* __restrict__ W, bf16* __restrict__ Wt, int K, int N)
{
    __shared__ float t[32][33];
    const int bx = blockIdx.x * 32;   // over N
    const int by = blockIdx.y * 32;   // over K
    const int tx = threadIdx.x & 31;
    const int ty = threadIdx.x >> 5;  // 0..7
    #pragma unroll
    for (int i = 0; i < 4; ++i)
        t[ty + i * 8][tx] = W[(long)(by + ty + i * 8) * N + bx + tx];
    __syncthreads();
    #pragma unroll
    for (int i = 0; i < 4; ++i)
        Wt[(long)(bx + ty + i * 8) * K + by + tx] = (bf16)t[tx][ty + i * 8];
}

// ---------------------------------------------------------------------------
// Masked softmax over 512 columns; writes bf16 probabilities.
// row -> b via rows_per_b (6144 for full attn, 12 for neighbor).
// ---------------------------------------------------------------------------
__global__ __launch_bounds__(256) void softmax_mask(
    const float* __restrict__ S, const int* __restrict__ mask,
    bf16* __restrict__ P, int rows_per_b)
{
    const int row = blockIdx.x;
    const int b   = row / rows_per_b;
    const int tid = threadIdx.x;
    const float* s  = S + (long)row * SQ;
    bf16*        p  = P + (long)row * SQ;
    const int*   mk = mask + b * SQ;
    __shared__ float red[256];

    float v0 = mk[tid]       ? s[tid]       : -__builtin_inff();
    float v1 = mk[tid + 256] ? s[tid + 256] : -__builtin_inff();

    red[tid] = fmaxf(v0, v1);
    __syncthreads();
    for (int o = 128; o > 0; o >>= 1) {
        if (tid < o) red[tid] = fmaxf(red[tid], red[tid + o]);
        __syncthreads();
    }
    float mx = red[0];
    __syncthreads();

    float e0 = (v0 > -__builtin_inff()) ? __expf(v0 - mx) : 0.0f;
    float e1 = (v1 > -__builtin_inff()) ? __expf(v1 - mx) : 0.0f;
    red[tid] = e0 + e1;
    __syncthreads();
    for (int o = 128; o > 0; o >>= 1) {
        if (tid < o) red[tid] += red[tid + o];
        __syncthreads();
    }
    float inv = red[0] > 0.0f ? 1.0f / red[0] : 0.0f;
    p[tid]       = (bf16)(e0 * inv);
    p[tid + 256] = (bf16)(e1 * inv);
}

// ---------------------------------------------------------------------------
// LayerNorm over D=768 with optional full residual and broadcast residual.
// Writes fp32 and (optionally) bf16 copies.
// ---------------------------------------------------------------------------
__global__ __launch_bounds__(256) void layernorm_k(
    const float* __restrict__ X, const float* __restrict__ addF,
    const float* __restrict__ addB, const float* __restrict__ g,
    const float* __restrict__ beta,
    float* __restrict__ outF, bf16* __restrict__ outB)
{
    const int row = blockIdx.x;
    const int b   = row >> 9;
    const int tid = threadIdx.x;
    const float* x = X + (long)row * DIM;
    float v[3];
    #pragma unroll
    for (int i = 0; i < 3; ++i) {
        int d = tid + i * 256;
        float t = x[d];
        if (addF) t += addF[(long)row * DIM + d];
        if (addB) t += addB[b * DIM + d];
        v[i] = t;
    }
    __shared__ float red[256];
    red[tid] = v[0] + v[1] + v[2];
    __syncthreads();
    for (int o = 128; o > 0; o >>= 1) {
        if (tid < o) red[tid] += red[tid + o];
        __syncthreads();
    }
    float mean = red[0] * (1.0f / DIM);
    __syncthreads();
    float q = 0.0f;
    #pragma unroll
    for (int i = 0; i < 3; ++i) { float d = v[i] - mean; q += d * d; }
    red[tid] = q;
    __syncthreads();
    for (int o = 128; o > 0; o >>= 1) {
        if (tid < o) red[tid] += red[tid + o];
        __syncthreads();
    }
    float rstd = rsqrtf(red[0] * (1.0f / DIM) + 1e-12f);
    #pragma unroll
    for (int i = 0; i < 3; ++i) {
        int d = tid + i * 256;
        float y = (v[i] - mean) * rstd * g[d] + beta[d];
        outF[(long)row * DIM + d] = y;
        if (outB) outB[(long)row * DIM + d] = (bf16)y;
    }
}

// ---------------------------------------------------------------------------
// Masked mean pooling: pooled[b,d] = sum_s X[b,s,d]*m[b,s] / sum_s m[b,s]
// ---------------------------------------------------------------------------
__global__ __launch_bounds__(256) void masked_mean_k(
    const float* __restrict__ X, const int* __restrict__ mask,
    float* __restrict__ pooled, bf16* __restrict__ pooledb)
{
    const int b = blockIdx.x;
    const int d = blockIdx.y * 256 + threadIdx.x;
    float s = 0.0f, c = 0.0f;
    for (int t = 0; t < SQ; ++t) {
        if (mask[b * SQ + t]) {
            s += X[((long)(b * SQ + t)) * DIM + d];
            c += 1.0f;
        }
    }
    float v = s / c;
    pooled[b * DIM + d]  = v;
    pooledb[b * DIM + d] = (bf16)v;
}

__global__ __launch_bounds__(256) void cvt_copy_k(
    const float* __restrict__ in, float* __restrict__ outF,
    bf16* __restrict__ outB, long n)
{
    long i = (long)blockIdx.x * 256 + threadIdx.x;
    if (i < n) { float v = in[i]; outF[i] = v; outB[i] = (bf16)v; }
}

__global__ __launch_bounds__(256) void copy_k(
    const float* __restrict__ in, float* __restrict__ out, long n)
{
    long i = (long)blockIdx.x * 256 + threadIdx.x;
    if (i < n) out[i] = in[i];
}

// ---------------------------------------------------------------------------
// Host driver
// ---------------------------------------------------------------------------
static const size_t kParamElems[26] = {
    (size_t)DIM * DIM, DIM,      //  0 Wq,  1 bq
    (size_t)DIM * DIM, DIM,      //  2 Wk,  3 bk
    (size_t)DIM * DIM, DIM,      //  4 Wv,  5 bv
    (size_t)DIM * DIM, DIM,      //  6 Wo,  7 bo
    (size_t)DIM * DIM, DIM,      //  8 nWq, 9 nbq
    (size_t)DIM * DIM, DIM,      // 10 nWk,11 nbk
    (size_t)DIM * DIM, DIM,      // 12 nWv,13 nbv
    (size_t)DIM * DIM, DIM,      // 14 nWo,15 nbo
    (size_t)DIM * HID, HID,      // 16 W1, 17 b1
    (size_t)HID * DIM, DIM,      // 18 W2, 19 b2
    DIM, DIM,                    // 20 ln1_w, 21 ln1_b
    DIM, DIM,                    // 22 ln2_w, 23 ln2_b
    DIM, DIM                     // 24 nln_w, 25 nln_b
};

static inline const float* prm(void* const* d_in, int strm, int idx, int layer) {
    return (const float*)d_in[2 + strm * 26 + idx] + (size_t)layer * kParamElems[idx];
}

static inline void launch_gemm(hipStream_t st, const bf16* A, const bf16* B,
    const float* bias, float* oF, bf16* oB,
    int M, int N, int K, int lda, int ldb, int ldo,
    long sA, long sB, long sC, float alpha, int act, int mode, int Z)
{
    dim3 g(N / 64, (M + 127) / 128, Z);
    gemm_nt<<<g, 256, 0, st>>>(A, B, bias, oF, oB, M, N, K, lda, ldb, ldo,
                               sA, sB, sC, alpha, act, mode);
}

static inline void launch_transp(hipStream_t st, const float* W, bf16* Wt, int K, int N) {
    transp_k<<<dim3(N / 32, K / 32), 256, 0, st>>>(W, Wt, K, N);
}

extern "C" void kernel_launch(void* const* d_in, const int* in_sizes, int n_in,
                              void* d_out, int out_size, void* d_ws, size_t ws_size,
                              hipStream_t stream)
{
    (void)in_sizes; (void)n_in; (void)out_size; (void)ws_size;
    const float* xin[2]  = { (const float*)d_in[0], (const float*)d_in[1] };
    const int*   mask[2] = { (const int*)d_in[54], (const int*)d_in[55] };

    // Workspace arena
    char* base = (char*)d_ws;
    size_t off = 0;
    auto alloc = [&](size_t bytes) -> void* {
        off = (off + 255) & ~(size_t)255;
        void* p = base + off; off += bytes; return p;
    };
    const size_t BSD = (size_t)BS * DIM;

    float *xf[2], *tmpf[2], *selff[2], *pooledf[2];
    bf16  *xb[2], *selfb[2], *pooledb[2];
    for (int s = 0; s < 2; ++s) {
        xf[s]    = (float*)alloc(BSD * 4);
        xb[s]    = (bf16*)alloc(BSD * 2);
        tmpf[s]  = (float*)alloc(BSD * 4);
        selff[s] = (float*)alloc(BSD * 4);
        selfb[s] = (bf16*)alloc(BSD * 2);
    }
    bf16*  Qb     = (bf16*)alloc(BSD * 2);
    bf16*  Kb     = (bf16*)alloc(BSD * 2);
    bf16*  Vtb    = (bf16*)alloc(BSD * 2);
    bf16*  ctxb   = (bf16*)alloc(BSD * 2);
    float* scores = (float*)alloc((size_t)BH * SQ * SQ * 4);
    bf16*  Pb     = (bf16*)alloc((size_t)BH * SQ * SQ * 2);
    bf16*  hb     = (bf16*)alloc((size_t)BS * HID * 2);
    bf16*  Wt     = (bf16*)alloc((size_t)HID * DIM * 2);
    for (int s = 0; s < 2; ++s) {
        pooledf[s] = (float*)alloc(NB * DIM * 4);
        pooledb[s] = (bf16*)alloc(NB * DIM * 2);
    }
    bf16*  nqb   = (bf16*)alloc(NB * DIM * 2);
    float* nscf  = (float*)alloc((size_t)BH * SQ * 4);
    bf16*  nPb   = (bf16*)alloc((size_t)BH * SQ * 2);
    bf16*  nctxb = (bf16*)alloc(NB * DIM * 2);
    float* nOf   = (float*)alloc(NB * DIM * 4);

    // Initialize activations (fp32 copy + bf16 shadow)
    for (int s = 0; s < 2; ++s)
        cvt_copy_k<<<(int)((BSD + 255) / 256), 256, 0, stream>>>(xin[s], xf[s], xb[s], (long)BSD);

    for (int l = 0; l < 6; ++l) {
        // ---------- self attention per stream ----------
        for (int s = 0; s < 2; ++s) {
            launch_transp(stream, prm(d_in, s, 0, l), Wt, DIM, DIM);
            launch_gemm(stream, xb[s], Wt, prm(d_in, s, 1, l), nullptr, Qb,
                        BS, DIM, DIM, DIM, DIM, 0, 0, 0, 0, 0.125f, 0, OUT_QK, 1);
            launch_transp(stream, prm(d_in, s, 2, l), Wt, DIM, DIM);
            launch_gemm(stream, xb[s], Wt, prm(d_in, s, 3, l), nullptr, Kb,
                        BS, DIM, DIM, DIM, DIM, 0, 0, 0, 0, 1.0f, 0, OUT_QK, 1);
            launch_transp(stream, prm(d_in, s, 4, l), Wt, DIM, DIM);
            launch_gemm(stream, xb[s], Wt, prm(d_in, s, 5, l), nullptr, Vtb,
                        BS, DIM, DIM, DIM, DIM, 0, 0, 0, 0, 1.0f, 0, OUT_VT, 1);
            // scores[z] = Q[z] (512x64) x K[z]^T (64x512), z = (b,h)
            launch_gemm(stream, Qb, Kb, nullptr, scores, nullptr,
                        SQ, SQ, DH, DH, DH, SQ,
                        (long)SQ * DH, (long)SQ * DH, (long)SQ * SQ, 1.0f, 0, OUT_F32, BH);
            softmax_mask<<<BH * SQ, 256, 0, stream>>>(scores, mask[s], Pb, NH * SQ);
            // ctx[z] = P[z] (512x512) x Vt[z]^T -> scatter to [BS x 768]
            launch_gemm(stream, Pb, Vtb, nullptr, nullptr, ctxb,
                        SQ, DH, SQ, SQ, SQ, 0,
                        (long)SQ * SQ, (long)DH * SQ, 0, 1.0f, 0, OUT_CTX, BH);
            launch_transp(stream, prm(d_in, s, 6, l), Wt, DIM, DIM);
            launch_gemm(stream, ctxb, Wt, prm(d_in, s, 7, l), tmpf[s], nullptr,
                        BS, DIM, DIM, DIM, DIM, DIM, 0, 0, 0, 1.0f, 0, OUT_F32, 1);
            layernorm_k<<<BS, 256, 0, stream>>>(tmpf[s], xf[s], nullptr,
                prm(d_in, s, 20, l), prm(d_in, s, 21, l), selff[s], selfb[s]);
        }
        // ---------- pooled queries ----------
        for (int s = 0; s < 2; ++s)
            masked_mean_k<<<dim3(NB, 3), 256, 0, stream>>>(selff[s], mask[s],
                                                           pooledf[s], pooledb[s]);
        // ---------- neighbor attention (single-row query per (b,h)) ----------
        for (int s = 0; s < 2; ++s) {
            const int o = 1 - s;
            launch_transp(stream, prm(d_in, s, 8, l), Wt, DIM, DIM);
            launch_gemm(stream, pooledb[o], Wt, prm(d_in, s, 9, l), nullptr, nqb,
                        NB, DIM, DIM, DIM, DIM, DIM, 0, 0, 0, 0.125f, 0, OUT_BF16, 1);
            launch_transp(stream, prm(d_in, s, 10, l), Wt, DIM, DIM);
            launch_gemm(stream, selfb[s], Wt, prm(d_in, s, 11, l), nullptr, Kb,
                        BS, DIM, DIM, DIM, DIM, 0, 0, 0, 0, 1.0f, 0, OUT_QK, 1);
            launch_transp(stream, prm(d_in, s, 12, l), Wt, DIM, DIM);
            launch_gemm(stream, selfb[s], Wt, prm(d_in, s, 13, l), nullptr, Vtb,
                        BS, DIM, DIM, DIM, DIM, 0, 0, 0, 0, 1.0f, 0, OUT_VT, 1);
            // nscores[z] (1x512) = nq[z] (1x64) x K[z]^T
            launch_gemm(stream, nqb, Kb, nullptr, nscf, nullptr,
                        1, SQ, DH, DH, DH, SQ,
                        (long)DH, (long)SQ * DH, (long)SQ, 1.0f, 0, OUT_F32, BH);
            softmax_mask<<<BH, 256, 0, stream>>>(nscf, mask[s], nPb, NH);
            // nctx[z] (1x64) = nP[z] (1x512) x Vt[z]^T
            launch_gemm(stream, nPb, Vtb, nullptr, nullptr, nctxb,
                        1, DH, SQ, SQ, SQ, DH,
                        (long)SQ, (long)DH * SQ, (long)DH, 1.0f, 0, OUT_BF16, BH);
            launch_transp(stream, prm(d_in, s, 14, l), Wt, DIM, DIM);
            launch_gemm(stream, nctxb, Wt, prm(d_in, s, 15, l), nOf, nullptr,
                        NB, DIM, DIM, DIM, DIM, DIM, 0, 0, 0, 1.0f, 0, OUT_F32, 1);
            layernorm_k<<<BS, 256, 0, stream>>>(selff[s], nullptr, nOf,
                prm(d_in, s, 24, l), prm(d_in, s, 25, l), xf[s], xb[s]);
        }
        // ---------- FFN per stream ----------
        for (int s = 0; s < 2; ++s) {
            launch_transp(stream, prm(d_in, s, 16, l), Wt, DIM, HID);
            launch_gemm(stream, xb[s], Wt, prm(d_in, s, 17, l), nullptr, hb,
                        BS, HID, DIM, DIM, DIM, HID, 0, 0, 0, 1.0f, 1, OUT_BF16, 1);
            launch_transp(stream, prm(d_in, s, 18, l), Wt, HID, DIM);
            launch_gemm(stream, hb, Wt, prm(d_in, s, 19, l), tmpf[s], nullptr,
                        BS, DIM, HID, HID, HID, DIM, 0, 0, 0, 1.0f, 0, OUT_F32, 1);
            layernorm_k<<<BS, 256, 0, stream>>>(tmpf[s], xf[s], nullptr,
                prm(d_in, s, 22, l), prm(d_in, s, 23, l), xf[s], xb[s]);
        }
    }

    for (int s = 0; s < 2; ++s)
        copy_k<<<(int)((BSD + 255) / 256), 256, 0, stream>>>(
            xf[s], (float*)d_out + (size_t)s * BSD, (long)BSD);
}
